// WavefrontEngine_38319698215501
// MI455X (gfx1250) — matrix operand order repaired and checked by reading, not compiled, not run on hardware
//
#include <hip/hip_runtime.h>
#include <math.h>

// Problem constants (from reference): L=8 layers, T=512 time, B=8 batch, D=512 dim
#define Lq 8
#define Tq 512
#define Bq 8
#define Dq 512
#define NSLOT 8          // ring-buffer depth per layer for bf16 inter-layer hand-off

#if __has_builtin(__builtin_amdgcn_global_load_async_to_lds_b128) && \
    __has_builtin(__builtin_amdgcn_s_wait_asynccnt)
#define HAVE_ASYNC_LDS 1
#else
#define HAVE_ASYNC_LDS 0
#endif

typedef __attribute__((ext_vector_type(16))) __bf16 v16bf;
typedef __attribute__((ext_vector_type(8)))  float  v8f;

#if HAVE_ASYNC_LDS
// exact pointee type expected by the async builtin (from hipcc diagnostic):
// 'int __vector(4) __device__ *' -> v4i in address space 1 / 3
typedef int v4i __attribute__((vector_size(16)));
typedef __attribute__((address_space(1))) v4i as1_v4i;
typedef __attribute__((address_space(3))) v4i as3_v4i;
#endif

__device__ __forceinline__ float tanh_fast(float v) {
#if __has_builtin(__builtin_amdgcn_tanhf)
  return __builtin_amdgcn_tanhf(v);      // hardware v_tanh_f32 (gfx1250 TRANS op)
#else
  return tanhf(v);
#endif
}

// ---------------------------------------------------------------------------
// Weight pre-swizzle: fp32 [3][L][D][D] row-major -> bf16 in WMMA B-fragment
// order. The B fragment (32x16) this code builds holds, per lane
// (lane = hi*16 + n), 16 bf16 = K = kchunk*32 + hi*16 + i (i=0..15), column n.
// Each lane's 16 elements are stored contiguously (32 B) so the main kernel
// loads one v16bf (2x global_load_b128) per fragment, coalesced.
// ---------------------------------------------------------------------------
__global__ void swizzle_weights(const float* __restrict__ Wx,
                                const float* __restrict__ Ws,
                                const float* __restrict__ Wo,
                                __bf16* __restrict__ wsw) {
  const int NFRAG = 3 * Lq * 16 * 32 * 32;          // 393216 fragments
  int f = blockIdx.x * blockDim.x + threadIdx.x;
  if (f >= NFRAG) return;
  int lane =  f        & 31;
  int nt   = (f >> 5)  & 31;
  int kc   = (f >> 10) & 15;
  int l    = (f >> 14) & 7;
  int mat  =  f >> 17;                               // 0=Wx 1=Ws 2=Wo
  const float* W = (mat == 0 ? Wx : (mat == 1 ? Ws : Wo)) + (size_t)l * Dq * Dq;
  const int hi    = lane >> 4;
  const int n     = nt * 16 + (lane & 15);
  const int kbase = kc * 32 + hi * 16;
  __bf16* dst = wsw + (size_t)(mat * Lq + l) * Dq * Dq
                    + (size_t)(kc * 32 + nt) * 512 + lane * 16;
#pragma unroll
  for (int i = 0; i < 16; ++i)
    dst[i] = (__bf16)W[(size_t)(kbase + i) * Dq + n];
}

__global__ void init_flags(unsigned int* flags) {
  if (threadIdx.x < 16) flags[threadIdx.x] = 0u;
}

// ---------------------------------------------------------------------------
// One K-sweep of C += A(16x512, LDS bf16) * B(512x512, swizzled bf16) for
// this wave's 4 N-tiles. A-fragment layout (16-bit A 16x32):
//   lane (hi*16+m), element i     -> K = kb + hi*16 + i (i=0..15)
// ---------------------------------------------------------------------------
__device__ __forceinline__ void gemm_acc(const __bf16 (*A)[Dq],
                                         const __bf16* __restrict__ Wsw,
                                         v8f c[4], int lane, int wave) {
  const int hi = lane >> 4;
  const int m  = lane & 15;
  const v16bf* Bp = (const v16bf*)Wsw;
  const __bf16* arow = &A[m][0];
#pragma unroll 1
  for (int kc = 0; kc < 16; ++kc) {
    const int kb = kc * 32;
    v16bf a;
#pragma unroll
    for (int i = 0; i < 8; ++i) a[i]     = arow[kb + hi * 16 + i];
#pragma unroll
    for (int i = 0; i < 8; ++i) a[8 + i] = arow[kb + hi * 16 + 8 + i];
#pragma unroll
    for (int j = 0; j < 4; ++j) {
      v16bf bfrag = Bp[(size_t)(kc * 32 + (wave * 4 + j)) * 32 + lane];
      c[j] = __builtin_amdgcn_wmma_f32_16x16x32_bf16(
                 false, a, false, bfrag, (short)0, c[j], false, false);
    }
  }
}

// ---------------------------------------------------------------------------
// Persistent pipelined wavefront: 8 resident workgroups, workgroup l owns
// layer l, marches t = 0..T-1.
//   h    = tanh(inp @ Wx[l] + state @ Ws[l] + b[l])  -> g1[cell], LDS (bf16)
//   out0 = h @ Wo[l]                                 -> g0[cell] (fp32) and a
//          bf16 ring slot for layer l+1.
// Recurrent state never leaves LDS. Inter-layer input arrives via
// GLOBAL_LOAD_ASYNC_TO_LDS_B128 issued BEFORE the state@Ws GEMM and waited
// after it (s_wait_asynccnt), hiding the hand-off latency behind 64 WMMAs.
// 256 threads = 8 wave32; wave w owns N columns [w*64, w*64+64).
// ---------------------------------------------------------------------------
__global__ __launch_bounds__(256)
void wavefront_persistent(const float* __restrict__ x,
                          const __bf16* __restrict__ wsw,
                          const float* __restrict__ bias,
                          float* g0, float* g1,
                          unsigned int* flags,      // [0..7]=prod, [8..15]=cons
                          __bf16* stage) {          // [7][NSLOT][Bq*Dq] bf16 ring
  __shared__ __align__(16) __bf16 sIn[16][Dq]; // depth input; rows 8..15 stay 0
  __shared__ __align__(16) __bf16 sSt[16][Dq]; // state h (bf16), also A of h@Wo

  const int l    = blockIdx.x;
  const int tid  = threadIdx.x;
  const int lane = tid & 31;
  const int wave = tid >> 5;
  const int hi   = lane >> 4;
  const int nlo  = lane & 15;
  unsigned int* prodF = flags;
  unsigned int* consF = flags + 8;

  // t=0 state is zeros; input pad rows zeroed once (never written again)
  for (int idx = tid; idx < 16 * Dq; idx += 256)
    sSt[idx >> 9][idx & (Dq - 1)] = (__bf16)0.0f;
  for (int idx = tid; idx < 8 * Dq; idx += 256)
    sIn[8 + (idx >> 9)][idx & (Dq - 1)] = (__bf16)0.0f;

  const __bf16* Wxs = wsw + (size_t)(0 * Lq + l) * Dq * Dq;
  const __bf16* Wss = wsw + (size_t)(1 * Lq + l) * Dq * Dq;
  const __bf16* Wos = wsw + (size_t)(2 * Lq + l) * Dq * Dq;

  float bb[4];
#pragma unroll
  for (int j = 0; j < 4; ++j)
    bb[j] = bias[(size_t)l * Dq + (wave * 4 + j) * 16 + nlo];

  for (int t = 0; t < Tq; ++t) {
    // ---- wait: producer data ready, and our outgoing ring slot free ----
    if (tid == 0) {
      if (l > 0)
        while (__hip_atomic_load(&prodF[l - 1], __ATOMIC_ACQUIRE,
                                 __HIP_MEMORY_SCOPE_AGENT) <= (unsigned)t)
          __builtin_amdgcn_s_sleep(2);
      if (l < Lq - 1 && t >= NSLOT)
        while (__hip_atomic_load(&consF[l], __ATOMIC_ACQUIRE,
                                 __HIP_MEMORY_SCOPE_AGENT) < (unsigned)(t - NSLOT + 1))
          __builtin_amdgcn_s_sleep(2);
    }
    __syncthreads();   // also covers the t=0 LDS init

    // ---- issue input staging (rows 0..7 of sIn) ----
    if (l == 0) {
      const float* xs = x + (size_t)t * Dq;             // x[m, t, :], stride T*D
      for (int idx = tid; idx < Bq * Dq; idx += 256) {
        int m = idx >> 9, k = idx & (Dq - 1);
        sIn[m][k] = (__bf16)xs[(size_t)m * Tq * Dq + k];
      }
    } else {
      const __bf16* src =
          stage + ((size_t)(l - 1) * NSLOT + (t & (NSLOT - 1))) * (Bq * Dq);
#if HAVE_ASYNC_LDS
      as1_v4i* gsrc = (as1_v4i*)((const char*)src + (size_t)tid * 16);
      as3_v4i* ldst = (as3_v4i*)((char*)&sIn[0][0] + (size_t)tid * 16);
      __builtin_amdgcn_global_load_async_to_lds_b128(gsrc, ldst, 0, 0);
      __builtin_amdgcn_global_load_async_to_lds_b128(gsrc, ldst, 4096, 0);
#else
      for (int idx = tid; idx < Bq * Dq; idx += 256)
        sIn[idx >> 9][idx & (Dq - 1)] = src[idx];
#endif
    }

    // ---- state @ Ws first: overlaps the async inter-layer copy ----
    v8f c[4] = {};
    gemm_acc(sSt, Wss, c, lane, wave);

    // ---- staging complete; free the consumed ring slot ----
#if HAVE_ASYNC_LDS
    if (l > 0) __builtin_amdgcn_s_wait_asynccnt(0);
#endif
    __syncthreads();
    if (l > 0 && tid == 0)
      __hip_atomic_store(&consF[l - 1], (unsigned)(t + 1), __ATOMIC_RELEASE,
                         __HIP_MEMORY_SCOPE_AGENT);

    // ---- += inp @ Wx ----
    gemm_acc(sIn, Wxs, c, lane, wave);

    // ---- h = tanh(.+b): refresh sSt in place, emit g1 rows < 8 ----
    // (all waves passed the mid barrier, so no one is still reading sSt)
    // C layout: VGPR r holds (m = r + 8*hi, n = tile*16 + nlo)
    const int cell = l * Tq + t;
    float* g1c = g1 + (size_t)cell * Bq * Dq;
#pragma unroll
    for (int j = 0; j < 4; ++j) {
      const int n = (wave * 4 + j) * 16 + nlo;
      float hv[8];
#pragma unroll
      for (int r = 0; r < 8; ++r) {
        hv[r] = tanh_fast(c[j][r] + bb[j]);
        sSt[r + 8 * hi][n] = (__bf16)hv[r];
      }
      if (hi == 0) {
#pragma unroll
        for (int r = 0; r < 8; ++r) g1c[(size_t)r * Dq + n] = hv[r];
      }
    }
    __syncthreads();

    // ---- out0 = h @ Wo ----
    v8f o[4] = {};
    gemm_acc(sSt, Wos, o, lane, wave);

    float* g0c = g0 + (size_t)cell * Bq * Dq;
    __bf16* sslot = stage + ((size_t)l * NSLOT + (t & (NSLOT - 1))) * (Bq * Dq);
#pragma unroll
    for (int j = 0; j < 4; ++j) {
      const int n = (wave * 4 + j) * 16 + nlo;
      if (hi == 0) {
#pragma unroll
        for (int r = 0; r < 8; ++r) {
          g0c[(size_t)r * Dq + n] = o[j][r];
          if (l < Lq - 1) sslot[r * Dq + n] = (__bf16)o[j][r];
        }
      }
    }

    // ---- publish cell (l,t) to layer l+1 ----
    if (l < Lq - 1) {
      __threadfence();     // order this thread's ring/g0 stores device-wide
      __syncthreads();     // all waves fenced before the flag release
      if (tid == 0)
        __hip_atomic_store(&prodF[l], (unsigned)(t + 1), __ATOMIC_RELEASE,
                           __HIP_MEMORY_SCOPE_AGENT);
    }
  }
}

// ---------------------------------------------------------------------------
// Host: swizzle weights into d_ws (12 MB bf16), zero flags, launch ONE
// persistent kernel with 8 co-resident workgroups. ws layout:
//   [0, 12 MiB)        : bf16 swizzled weights (3*L*D*D)
//   [12 MiB, +64 B)    : 16 flags
//   [+256 B, ...)      : bf16 ring buffers, 7 layers * 8 slots * 8 KiB = 448 KiB
// ---------------------------------------------------------------------------
extern "C" void kernel_launch(void* const* d_in, const int* in_sizes, int n_in,
                              void* d_out, int out_size, void* d_ws, size_t ws_size,
                              hipStream_t stream) {
  (void)in_sizes; (void)n_in; (void)out_size; (void)ws_size;
  const float* x  = (const float*)d_in[0];
  const float* Wx = (const float*)d_in[1];
  const float* Ws = (const float*)d_in[2];
  const float* Wo = (const float*)d_in[3];
  const float* b  = (const float*)d_in[4];

  float* g0 = (float*)d_out;                                   // [L*T, B, D]
  float* g1 = (float*)d_out + (size_t)Lq * Tq * Bq * Dq;       // [L*T, B, D]

  __bf16* wsw = (__bf16*)d_ws;
  const size_t wbytes = (size_t)3 * Lq * Dq * Dq * sizeof(__bf16);
  unsigned int* flags = (unsigned int*)((char*)d_ws + wbytes);
  __bf16* stage = (__bf16*)((char*)d_ws + wbytes + 256);

  const int NFRAG = 3 * Lq * 16 * 32 * 32;
  swizzle_weights<<<(NFRAG + 255) / 256, 256, 0, stream>>>(Wx, Ws, Wo, wsw);
  init_flags<<<1, 32, 0, stream>>>(flags);
  wavefront_persistent<<<Lq, 256, 0, stream>>>(x, wsw, b, g0, g1, flags, stage);
}
